// PointNetPlusPlus_5016521802587
// MI455X (gfx1250) — compile-verified, matrix operand order checked by reading, not run on hardware
//
#include <hip/hip_runtime.h>
#include <stdint.h>

// ---------------------------------------------------------------------------
// PointNet++ forward for MI455X (gfx1250, wave32, WMMA).
// All GEMM-shaped work runs on v_wmma_f32_16x16x32_f16 (f16 in, f32 acc).
// Neighbor selection (knn / deterministic ball query) is VALU + LDS tiling.
// ---------------------------------------------------------------------------

typedef __attribute__((ext_vector_type(16))) _Float16 v16h;
typedef __attribute__((ext_vector_type(8)))  float    v8f;

union FragH { v16h v; uint32_t u[8]; };

// K index held by VGPR v (0..7), lane-half h, for 16-bit A/B fragments
// (per CDNA5 ISA 7.12.2, 16-bit A-Matrix 16x32 table).
__device__ __forceinline__ int kmap(int v, int h) {
  return (v < 4) ? (v * 2 + h * 8) : (16 + (v - 4) * 2 + h * 8);
}

// Load one 16x32 f16 fragment for this lane from a row pointer (row = lane%16
// already applied by caller). Pairs (K,K+1) are contiguous => 4-byte loads.
__device__ __forceinline__ v16h load_frag_row(const _Float16* rowPtr, int h) {
  FragH f;
#pragma unroll
  for (int v = 0; v < 8; ++v)
    f.u[v] = *(const uint32_t*)(rowPtr + kmap(v, h));
  return f.v;
}

__device__ __forceinline__ v8f wmma_f16(v16h a, v16h b, v8f c) {
  return __builtin_amdgcn_wmma_f32_16x16x32_f16(false, a, false, b, (short)0, c,
                                                false, false);
}

// Store D-fragment to LDS (f16) with bias + relu. D layout: n = lane%16,
// m = r + 8*(lane/16).
__device__ __forceinline__ void store_d_relu_lds(_Float16* Hbuf, int ld, v8f acc,
                                                 const float* bias, int mBase,
                                                 int nBase) {
  int lane = threadIdx.x & 31, h = lane >> 4, nn = lane & 15;
  float bb = bias[nBase + nn];
#pragma unroll
  for (int r = 0; r < 8; ++r) {
    float vv = acc[r] + bb;
    vv = vv > 0.f ? vv : 0.f;
    Hbuf[(size_t)(mBase + r + 8 * h) * ld + nBase + nn] = (_Float16)vv;
  }
}

// Second MLP layer (HxH) from LDS hidden buffer, with on-the-fly mean over the
// S neighbor rows folded into the D-fragment reduction. Writes f32 (optional)
// and f16 outputs for row n (lanes of half 0).
template <int S, int H>
__device__ __forceinline__ void gemm2_mean(const _Float16* H1,
                                           const _Float16* wT, const float* bias,
                                           float* outF, _Float16* outH) {
  int lane = threadIdx.x & 31, h = lane >> 4, nn = lane & 15;
  float csum[H / 16];
#pragma unroll
  for (int i = 0; i < H / 16; ++i) csum[i] = 0.f;

  for (int mT = 0; mT < S / 16; ++mT) {
    v16h aF[H / 32];
    const _Float16* aRow = H1 + (size_t)(mT * 16 + nn) * H;
#pragma unroll
    for (int kk = 0; kk < H / 32; ++kk) aF[kk] = load_frag_row(aRow + kk * 32, h);
#pragma unroll
    for (int nT = 0; nT < H / 16; ++nT) {
      v8f acc = {};
      const _Float16* wRow = wT + (size_t)(nT * 16 + nn) * H;
#pragma unroll
      for (int kk = 0; kk < H / 32; ++kk) {
        v16h bF = load_frag_row(wRow + kk * 32, h);
        acc = wmma_f16(aF[kk], bF, acc);
      }
      float bb = bias[nT * 16 + nn];
#pragma unroll
      for (int r = 0; r < 8; ++r) {
        float vv = acc[r] + bb;
        csum[nT] += vv > 0.f ? vv : 0.f;
      }
    }
  }
#pragma unroll
  for (int nT = 0; nT < H / 16; ++nT) {
    float c = csum[nT] + __shfl_xor(csum[nT], 16, 32);
    float mean = c * (1.f / (float)S);
    if (h == 0) {
      if (outF) outF[nT * 16 + nn] = mean;
      outH[nT * 16 + nn] = (_Float16)mean;
    }
  }
}

// ---------------------------------------------------------------------------
// Weight prep: f32 [C][D] -> f16 transposed [D][C]
// ---------------------------------------------------------------------------
__global__ void wtrans(const float* w, _Float16* wt, int Cd, int Dd) {
  int t = blockIdx.x * blockDim.x + threadIdx.x;
  if (t >= Cd * Dd) return;
  int d = t / Cd, c = t - d * Cd;
  wt[t] = (_Float16)w[(size_t)c * Dd + d];
}

// ---------------------------------------------------------------------------
// knn (k=32) on 2-D points; stable top-k insertion == argsort semantics.
// ---------------------------------------------------------------------------
__global__ void __launch_bounds__(128) knn32(const float* x, int* idxOut) {
  const int N = 2048, K = 32, T = 128, TC = 256;
  int b = blockIdx.y;
  int n = blockIdx.x * T + threadIdx.x;
  extern __shared__ char sm[];
  float* cand = (float*)sm;                                   // TC*2 floats
  float* dl = (float*)(sm + TC * 2 * 4) + threadIdx.x * K;    // per-thread dists
  int* il = (int*)(sm + TC * 2 * 4 + T * K * 4) + threadIdx.x * K;
  const float* xb = x + (size_t)b * N * 2;
  float qx = xb[n * 2], qy = xb[n * 2 + 1];
  int cnt = 0;
  for (int t0 = 0; t0 < N; t0 += TC) {
    __syncthreads();
    for (int i = threadIdx.x; i < TC * 2; i += T) cand[i] = xb[t0 * 2 + i];
    __syncthreads();
    for (int j = 0; j < TC; ++j) {
      float dx = qx - cand[2 * j], dy = qy - cand[2 * j + 1];
      float d2 = dx * dx + dy * dy;
      if (cnt == K && d2 >= dl[K - 1]) continue;
      int pos = cnt < K ? cnt : K - 1;
      while (pos > 0 && dl[pos - 1] > d2) {
        dl[pos] = dl[pos - 1];
        il[pos] = il[pos - 1];
        --pos;
      }
      dl[pos] = d2;
      il[pos] = t0 + j;
      if (cnt < K) ++cnt;
    }
  }
  int* op = idxOut + ((size_t)b * N + n) * K;
  for (int s = 0; s < K; ++s) op[s] = il[s];
}

// ---------------------------------------------------------------------------
// Deterministic ball query: ascending-index valid neighbors, cyclic repeat.
// ---------------------------------------------------------------------------
template <int C, int S, int T, int TC>
__global__ void __launch_bounds__(128) ball_kernel(const float* xf, int* idxOut,
                                                   float r2) {
  const int N = 2048;
  int b = blockIdx.y;
  int n = blockIdx.x * T + threadIdx.x;
  extern __shared__ char sm[];
  float* cand = (float*)sm;                                  // TC*C floats
  int* list = (int*)(sm + (size_t)TC * C * 4) + threadIdx.x * S;
  float q[C];
  const float* qp = xf + ((size_t)b * N + n) * C;
#pragma unroll
  for (int c = 0; c < C; ++c) q[c] = qp[c];
  int cnt = 0;
  for (int t0 = 0; t0 < N; t0 += TC) {
    __syncthreads();
    for (int i = threadIdx.x; i < TC * C; i += T)
      cand[i] = xf[((size_t)b * N + t0) * C + i];
    __syncthreads();
    for (int j = 0; j < TC; ++j) {
      float d2 = 0.f;
#pragma unroll
      for (int c = 0; c < C; ++c) {
        float df = q[c] - cand[j * C + c];
        d2 = fmaf(df, df, d2);
      }
      if (d2 <= r2) {
        if (cnt < S) list[cnt] = t0 + j;
        ++cnt;
      }
    }
  }
  int m = cnt < S ? cnt : S;  // >=1 (self always in ball)
  int* op = idxOut + ((size_t)b * N + n) * S;
  for (int s = 0; s < S; ++s) op[s] = list[s % m];
}

// ---------------------------------------------------------------------------
// SA1: gather 2-D coords, layer1 (2->64) in VALU, layer2 (64->64) WMMA + mean.
// One wave per point.
// ---------------------------------------------------------------------------
__global__ void __launch_bounds__(128) sa1_kernel(const float* x, const int* idx,
                                                  const float* w0,
                                                  const float* b0,
                                                  const _Float16* wT1,
                                                  const float* b1, float* x1f,
                                                  _Float16* x1h) {
  const int S = 32, H = 64, N = 2048;
  int WPB = blockDim.x >> 5, wv = threadIdx.x >> 5;
  int p = blockIdx.x * WPB + wv;
  int b = p >> 11;
  extern __shared__ char sm[];
  float2* gxy = (float2*)sm + wv * S;
  _Float16* H1 = (_Float16*)(sm + (size_t)WPB * S * 8) + (size_t)wv * S * H;
  int lane = threadIdx.x & 31;
  {
    int j = idx[(size_t)p * S + lane];
    const float* xp = x + ((size_t)b * N + j) * 2;
    gxy[lane] = make_float2(xp[0], xp[1]);
  }
  __syncthreads();
  for (int t = lane; t < S * H; t += 32) {
    int s = t >> 6, d = t & 63;
    float2 g = gxy[s];
    float v = g.x * w0[d] + g.y * w0[64 + d] + b0[d];
    H1[t] = (_Float16)(v > 0.f ? v : 0.f);
  }
  __syncthreads();
  gemm2_mean<S, H>(H1, wT1, b1, x1f + (size_t)p * H, x1h + (size_t)p * H);
}

// ---------------------------------------------------------------------------
// Fused set-abstraction MLP: gather(S neighbors, C feats) -> WMMA (C->H) relu
// -> LDS -> WMMA (H->H) relu -> mean over S. One wave per point.
// ---------------------------------------------------------------------------
template <int S, int C, int H>
__global__ void __launch_bounds__(128) sa_mlp(const _Float16* xh, const int* idx,
                                              const _Float16* wT0,
                                              const float* b0,
                                              const _Float16* wT1,
                                              const float* b1, float* outF,
                                              _Float16* outH, int ldOut,
                                              int colOff) {
  const int N = 2048;
  int WPB = blockDim.x >> 5, wv = threadIdx.x >> 5;
  int p = blockIdx.x * WPB + wv;
  int b = p >> 11;
  extern __shared__ char sm[];
  int* lidx = (int*)sm + wv * S;
  _Float16* H1 = (_Float16*)(sm + (size_t)WPB * S * 4) + (size_t)wv * S * H;
  int lane = threadIdx.x & 31, h = lane >> 4, nn = lane & 15;
  for (int i = lane; i < S; i += 32) lidx[i] = idx[(size_t)p * S + i];
  __syncthreads();
  const _Float16* xb = xh + (size_t)b * N * C;

  // Layer 1: (S x C) x (C x H), gathered A, bias+relu -> LDS
  for (int mT = 0; mT < S / 16; ++mT) {
    int row = lidx[mT * 16 + nn];
    const _Float16* aRow = xb + (size_t)row * C;
    v16h aF[C / 32];
#pragma unroll
    for (int kk = 0; kk < C / 32; ++kk) aF[kk] = load_frag_row(aRow + kk * 32, h);
    for (int nT = 0; nT < H / 16; ++nT) {
      v8f acc = {};
      const _Float16* wRow = wT0 + (size_t)(nT * 16 + nn) * C;
#pragma unroll
      for (int kk = 0; kk < C / 32; ++kk) {
        v16h bF = load_frag_row(wRow + kk * 32, h);
        acc = wmma_f16(aF[kk], bF, acc);
      }
      store_d_relu_lds(H1, H, acc, b0, mT * 16, nT * 16);
    }
  }
  __syncthreads();

  // Layer 2 (H->H) + mean over neighbors
  float* oF = outF ? outF + (size_t)p * H : nullptr;
  _Float16* oH = outH + (size_t)p * ldOut + colOff;
  gemm2_mean<S, H>(H1, wT1, b1, oF, oH);
}

// ---------------------------------------------------------------------------
// Generic dense layer: Out = relu(A(f16, RxK) * W^T(f16, DxK) + bias), f16 out
// written into a (possibly wider) buffer at colOff. One 16x16 tile per wave.
// ---------------------------------------------------------------------------
__global__ void __launch_bounds__(128) gemm_relu(const _Float16* A, int ldA,
                                                 const _Float16* WT,
                                                 const float* bias,
                                                 _Float16* Out, int ldOut,
                                                 int colOff, int R, int K,
                                                 int D) {
  int wv = threadIdx.x >> 5;
  int tile = blockIdx.x * (blockDim.x >> 5) + wv;
  int nTiles = D >> 4;
  int rT = tile / nTiles, cT = tile - rT * nTiles;
  if (rT >= (R >> 4)) return;
  int lane = threadIdx.x & 31, h = lane >> 4, nn = lane & 15;
  int n = cT * 16 + nn;
  v8f acc = {};
  const _Float16* aRow = A + (size_t)(rT * 16 + nn) * ldA;
  const _Float16* wRow = WT + (size_t)n * K;
  for (int kB = 0; kB < K; kB += 32) {
    __builtin_prefetch(wRow + kB + 128, 0, 0);  // -> global_prefetch_b8
    v16h fa = load_frag_row(aRow + kB, h);
    v16h fb = load_frag_row(wRow + kB, h);
    acc = wmma_f16(fa, fb, acc);
  }
  float bb = bias[n];
  _Float16* oBase = Out + colOff + n;
#pragma unroll
  for (int r = 0; r < 8; ++r) {
    float v = acc[r] + bb;
    v = v > 0.f ? v : 0.f;
    oBase[(size_t)(rT * 16 + r + 8 * h) * ldOut] = (_Float16)v;
  }
}

// ---------------------------------------------------------------------------
// Final 128->1 + sigmoid, f32 output.
// ---------------------------------------------------------------------------
__global__ void out_sigmoid(const _Float16* V, const float* ow, const float* ob,
                            float* out, int R) {
  int r = blockIdx.x * blockDim.x + threadIdx.x;
  if (r >= R) return;
  float acc = ob[0];
#pragma unroll 16
  for (int d = 0; d < 128; ++d) acc += (float)V[(size_t)r * 128 + d] * ow[d];
  out[r] = 1.0f / (1.0f + expf(-acc));
}

// ---------------------------------------------------------------------------
extern "C" void kernel_launch(void* const* d_in, const int* in_sizes, int n_in,
                              void* d_out, int out_size, void* d_ws,
                              size_t ws_size, hipStream_t stream) {
  (void)in_sizes; (void)n_in; (void)out_size; (void)ws_size;
  const int B = 8, N = 2048, RN = B * N;

  const float* x      = (const float*)d_in[0];
  const float* sa1_w0 = (const float*)d_in[1];
  const float* sa1_b0 = (const float*)d_in[2];
  const float* sa1_w1 = (const float*)d_in[3];
  const float* sa1_b1 = (const float*)d_in[4];
  const float* sa2_w0 = (const float*)d_in[5];
  const float* sa2_b0 = (const float*)d_in[6];
  const float* sa2_w1 = (const float*)d_in[7];
  const float* sa2_b1 = (const float*)d_in[8];
  const float* sa3_w0 = (const float*)d_in[9];
  const float* sa3_b0 = (const float*)d_in[10];
  const float* sa3_w1 = (const float*)d_in[11];
  const float* sa3_b1 = (const float*)d_in[12];
  const float* fp1_w0 = (const float*)d_in[13];
  const float* fp1_b0 = (const float*)d_in[14];
  const float* fp1_w1 = (const float*)d_in[15];
  const float* fp1_b1 = (const float*)d_in[16];
  const float* fp2_w0 = (const float*)d_in[17];
  const float* fp2_b0 = (const float*)d_in[18];
  const float* fp2_w1 = (const float*)d_in[19];
  const float* fp2_b1 = (const float*)d_in[20];
  const float* fc1_w  = (const float*)d_in[21];
  const float* fc1_b  = (const float*)d_in[22];
  const float* fc2_w  = (const float*)d_in[23];
  const float* fc2_b  = (const float*)d_in[24];
  const float* out_w  = (const float*)d_in[25];
  const float* out_b  = (const float*)d_in[26];

  char* ws = (char*)d_ws;
  size_t off = 0;
  auto alloc = [&](size_t bytes) -> void* {
    void* p = ws + off;
    off = (off + bytes + 255) & ~(size_t)255;
    return p;
  };

  // f16 transposed weights [D][K]
  _Float16* h_sa1w1 = (_Float16*)alloc((size_t)64 * 64 * 2);
  _Float16* h_sa2w0 = (_Float16*)alloc((size_t)128 * 64 * 2);
  _Float16* h_sa2w1 = (_Float16*)alloc((size_t)128 * 128 * 2);
  _Float16* h_sa3w0 = (_Float16*)alloc((size_t)256 * 128 * 2);
  _Float16* h_sa3w1 = (_Float16*)alloc((size_t)256 * 256 * 2);
  _Float16* h_fp1w0 = (_Float16*)alloc((size_t)512 * 128 * 2);
  _Float16* h_fp1w1 = (_Float16*)alloc((size_t)512 * 512 * 2);
  _Float16* h_fp2w0 = (_Float16*)alloc((size_t)256 * 64 * 2);
  _Float16* h_fp2w1 = (_Float16*)alloc((size_t)256 * 256 * 2);
  _Float16* h_fc1   = (_Float16*)alloc((size_t)256 * 1024 * 2);
  _Float16* h_fc2   = (_Float16*)alloc((size_t)128 * 256 * 2);
  // intermediates
  int*      idx1 = (int*)alloc((size_t)RN * 32 * 4);
  int*      idx2 = (int*)alloc((size_t)RN * 32 * 4);
  int*      idx3 = (int*)alloc((size_t)RN * 64 * 4);
  float*    x1f  = (float*)alloc((size_t)RN * 64 * 4);
  _Float16* x1h  = (_Float16*)alloc((size_t)RN * 64 * 2);
  float*    x2f  = (float*)alloc((size_t)RN * 128 * 4);
  _Float16* x2h  = (_Float16*)alloc((size_t)RN * 128 * 2);
  _Float16* cat  = (_Float16*)alloc((size_t)RN * 1024 * 2);  // [g | fp1h | x3]
  _Float16* t512 = (_Float16*)alloc((size_t)RN * 512 * 2);
  _Float16* t256 = (_Float16*)alloc((size_t)RN * 256 * 2);
  _Float16* u256 = (_Float16*)alloc((size_t)RN * 256 * 2);
  _Float16* v128 = (_Float16*)alloc((size_t)RN * 128 * 2);

  auto TW = [&](const float* w, _Float16* wt, int Cd, int Dd) {
    int tot = Cd * Dd;
    wtrans<<<(tot + 255) / 256, 256, 0, stream>>>(w, wt, Cd, Dd);
  };
  TW(sa1_w1, h_sa1w1, 64, 64);
  TW(sa2_w0, h_sa2w0, 64, 128);
  TW(sa2_w1, h_sa2w1, 128, 128);
  TW(sa3_w0, h_sa3w0, 128, 256);
  TW(sa3_w1, h_sa3w1, 256, 256);
  TW(fp1_w0, h_fp1w0, 128, 512);
  TW(fp1_w1, h_fp1w1, 512, 512);
  TW(fp2_w0, h_fp2w0, 64, 256);
  TW(fp2_w1, h_fp2w1, 256, 256);
  TW(fc1_w, h_fc1, 1024, 256);
  TW(fc2_w, h_fc2, 256, 128);

  // SA1: knn(32) on raw 2-D points, MLP [64,64], mean
  knn32<<<dim3(N / 128, B), 128, 256 * 2 * 4 + 128 * 32 * 8, stream>>>(x, idx1);
  sa1_kernel<<<RN / 4, 128, 4 * (32 * 8 + 32 * 64 * 2), stream>>>(
      x, idx1, sa1_w0, sa1_b0, h_sa1w1, sa1_b1, x1f, x1h);

  // SA2: ball(r=0.2, 32) on x1, MLP [128,128], mean
  ball_kernel<64, 32, 128, 64>
      <<<dim3(N / 128, B), 128, 64 * 64 * 4 + 128 * 32 * 4, stream>>>(
          x1f, idx2, 0.04f);
  sa_mlp<32, 64, 128><<<RN / 4, 128, 4 * (32 * 4 + 32 * 128 * 2), stream>>>(
      x1h, idx2, h_sa2w0, sa2_b0, h_sa2w1, sa2_b1, x2f, x2h, 128, 0);

  // SA3: ball(r=0.4, 64) on x2, MLP [256,256], mean -> cat[:, 768:1024]
  ball_kernel<128, 64, 64, 32>
      <<<dim3(N / 64, B), 64, 32 * 128 * 4 + 64 * 64 * 4, stream>>>(
          x2f, idx3, 0.16f);
  sa_mlp<64, 128, 256><<<RN, 32, 64 * 4 + 64 * 256 * 2, stream>>>(
      x2h, idx3, h_sa3w0, sa3_b0, h_sa3w1, sa3_b1, nullptr, cat, 1024, 768);

  // Dense tail (all WMMA); concat is just column offsets in `cat`.
  auto G = [&](const _Float16* A, int ldA, const _Float16* WT, const float* bias,
               _Float16* Out, int ldOut, int colOff, int K, int D) {
    int tiles = (RN / 16) * (D / 16);
    gemm_relu<<<tiles / 4, 128, 0, stream>>>(A, ldA, WT, bias, Out, ldOut,
                                             colOff, RN, K, D);
  };
  G(x2h, 128, h_fp1w0, fp1_b0, t512, 512, 0, 128, 512);      // fp1 layer 0
  G(t512, 512, h_fp1w1, fp1_b1, cat, 1024, 256, 512, 512);   // fp1 layer 1 -> cat
  G(x1h, 64, h_fp2w0, fp2_b0, t256, 256, 0, 64, 256);        // fp2 layer 0
  G(t256, 256, h_fp2w1, fp2_b1, cat, 1024, 0, 256, 256);     // fp2 layer 1 -> cat
  G(cat, 1024, h_fc1, fc1_b, u256, 256, 0, 1024, 256);       // fc1
  G(u256, 256, h_fc2, fc2_b, v128, 128, 0, 256, 128);        // fc2

  out_sigmoid<<<RN / 256, 256, 0, stream>>>(v128, out_w, out_b, (float*)d_out,
                                            RN);
}
